// NonLocalBlock_29334626631792
// MI455X (gfx1250) — matrix operand order repaired and checked by reading, not compiled, not run on hardware
//
#include <hip/hip_runtime.h>
#include <hip/hip_bf16.h>
#include <math.h>

#define B_   4
#define C_   256
#define D_   128
#define N_   4096
#define EPS_ 1e-5f

typedef __attribute__((ext_vector_type(16))) __bf16 v16bf;
typedef __attribute__((ext_vector_type(8)))  float  v8f;

union BFrag {
    v16bf          v;
    unsigned short u[16];
    unsigned int   w[8];
};

// Hardware bf16 convert (fptrunc -> v_cvt_pk_bf16_f32 on gfx1250).
__device__ __forceinline__ unsigned short f2bf(float f) {
    union { __bf16 h; unsigned short u; } cv;
    cv.h = (__bf16)f;
    return cv.u;
}

__device__ __forceinline__ v8f wmma_bf16(const BFrag& a, const BFrag& b, v8f c) {
    // D = A(16x32 bf16) * B(32x16 bf16) + C(16x16 f32)
    return __builtin_amdgcn_wmma_f32_16x16x32_bf16(
        /*neg_a=*/false, a.v, /*neg_b=*/false, b.v,
        /*c_mod=*/(short)0, c, /*reuse_a=*/false, /*reuse_b=*/false);
}

__device__ __forceinline__ v8f vzero8() {
    v8f z = {0.f, 0.f, 0.f, 0.f, 0.f, 0.f, 0.f, 0.f};
    return z;
}

// Async global -> LDS copy (ASYNCcnt-tracked, CDNA5 ISA).
// lds_off: LDS byte offset (low 32 bits of a generic shared pointer).
__device__ __forceinline__ void async_b128(unsigned int lds_off, const void* gaddr) {
    asm volatile("global_load_async_to_lds_b128 %0, %1, off"
                 :: "v"(lds_off), "v"(gaddr) : "memory");
}
#define WAIT_ASYNCCNT(n) asm volatile("s_wait_asynccnt " #n ::: "memory")

// -----------------------------------------------------------------------------
// Prep A: convert a weight matrix fp32 -> bf16 (same layout). Tiny, one-shot.
// -----------------------------------------------------------------------------
__global__ __launch_bounds__(256)
void cvtw_kernel(const float* __restrict__ src, unsigned short* __restrict__ dst,
                 int count)
{
    const int i = blockIdx.x * 256 + threadIdx.x;
    if (i < count) dst[i] = f2bf(src[i]);
}

// -----------------------------------------------------------------------------
// Prep B: x [B][C][N] fp32 -> xT [B][N][C] bf16 (projection B-fragment layout).
// -----------------------------------------------------------------------------
__global__ __launch_bounds__(256)
void cvtx_kernel(const float* __restrict__ x, unsigned short* __restrict__ xT)
{
    const size_t idx = (size_t)blockIdx.x * 256 + threadIdx.x; // over B*C*N
    const int n = (int)(idx % N_);
    const int c = (int)((idx / N_) % C_);
    const int b = (int)(idx / ((size_t)C_ * N_));
    xT[((size_t)b * N_ + n) * C_ + c] = f2bf(x[idx]);
}

// -----------------------------------------------------------------------------
// Kernel 1: 1x1-conv projections g/theta/phi (M=128, K=256, N=4096 per batch).
// gT stored [B][D][N] bf16; thetaT/phiT stored [B][N][D] bf16.
// -----------------------------------------------------------------------------
__global__ __launch_bounds__(256)
void proj_kernel(const unsigned short* __restrict__ xT,   // [B][N][C] bf16
                 const unsigned short* __restrict__ Wbf,  // [D][C] bf16
                 const float* __restrict__ bias,
                 unsigned short* __restrict__ outG,       // [B][D][N] (proj==0)
                 unsigned short* __restrict__ outR,       // [B][N][D] (proj!=0)
                 int proj)
{
    const int lane  = threadIdx.x & 31;
    const int wv    = threadIdx.x >> 5;
    const int hi    = lane >> 4;
    const int ln    = lane & 15;
    const int b     = blockIdx.y;
    const int nbase = blockIdx.x * 128 + wv * 16;
    const int n     = nbase + ln;

    v8f acc[8];
    #pragma unroll
    for (int i = 0; i < 8; ++i) acc[i] = vzero8();

    for (int kk = 0; kk < 8; ++kk) {                 // K = 256, step 32
        BFrag xb;
        #pragma unroll
        for (int v = 0; v < 8; ++v) {
            const int c0 = kk * 32 + (hi ? 8 : 0) + ((v >= 4) ? 16 : 0) + 2 * (v & 3);
            xb.w[v] = *(const unsigned int*)(xT + (((size_t)b * N_ + n) * C_ + c0));
        }
        #pragma unroll
        for (int dt = 0; dt < 8; ++dt) {             // 8 row tiles -> d = 0..127
            BFrag wa;
            const int drow = dt * 16 + ln;
            #pragma unroll
            for (int v = 0; v < 8; ++v) {
                const int kd = kk * 32 + (hi ? 8 : 0) + ((v >= 4) ? 16 : 0) + 2 * (v & 3);
                wa.w[v] = *(const unsigned int*)(Wbf + (size_t)drow * C_ + kd);
            }
            acc[dt] = wmma_bf16(wa, xb, acc[dt]);
        }
    }

    #pragma unroll
    for (int dt = 0; dt < 8; ++dt) {
        #pragma unroll
        for (int v = 0; v < 8; ++v) {
            const int d = dt * 16 + v + (hi ? 8 : 0);
            const unsigned short bf = f2bf(acc[dt][v] + bias[d]);
            if (proj == 0) outG[((size_t)b * D_ + d) * N_ + n] = bf;
            else           outR[((size_t)b * N_ + n) * D_ + d] = bf;
        }
    }
}

// -----------------------------------------------------------------------------
// Kernel 2: fused flash attention. Per block: 128 query rows (16 per wave),
// column loop in steps of 32 with online softmax. Shared phi/G tiles are
// staged into LDS with double-buffered async global->LDS copies.
// -----------------------------------------------------------------------------
__global__ __launch_bounds__(256)
void attn_kernel(const unsigned short* __restrict__ thetaT, // [B][N][D]
                 const unsigned short* __restrict__ phiT,   // [B][N][D]
                 const unsigned short* __restrict__ gT,     // [B][D][N]
                 unsigned short* __restrict__ ybf)          // [B][N][D]
{
    __shared__ __align__(16) unsigned short PhiS[2][32][128]; // [buf][m][d]
    __shared__ __align__(16) unsigned short GS[2][128][32];   // [buf][d][m]
    __shared__ __align__(16) unsigned short Plds[8][16][32];

    const int tid   = threadIdx.x;
    const int lane  = tid & 31;
    const int wv    = tid >> 5;
    const int hi    = lane >> 4;
    const int ln    = lane & 15;
    const int b     = blockIdx.y;
    const int nbase = blockIdx.x * 128 + wv * 16;

    // Stage phi (32 rows m x 128 d) and G (128 d x 32 m) tiles for column
    // block starting at mbase into LDS buffer `buf`: 4 async b128 per thread.
    auto stage = [&](int buf, int mbase) {
        #pragma unroll
        for (int i = 0; i < 2; ++i) {
            const int c  = tid + 256 * i;         // 512 16B chunks
            const int m  = c >> 4;
            const int dd = (c & 15) * 8;
            async_b128((unsigned int)(size_t)&PhiS[buf][m][dd],
                       phiT + (((size_t)b * N_ + mbase + m) * D_ + dd));
        }
        #pragma unroll
        for (int i = 0; i < 2; ++i) {
            const int c  = tid + 256 * i;
            const int d  = c >> 2;
            const int mm = (c & 3) * 8;
            async_b128((unsigned int)(size_t)&GS[buf][d][mm],
                       gT + (((size_t)b * D_ + d) * N_ + mbase + mm));
        }
    };

    // Resident theta A-fragments: 16 rows x K=128 (4 k-steps of 32)
    BFrag thA[4];
    #pragma unroll
    for (int kk = 0; kk < 4; ++kk) {
        #pragma unroll
        for (int v = 0; v < 8; ++v) {
            const int d = kk * 32 + (hi ? 8 : 0) + ((v >= 4) ? 16 : 0) + 2 * (v & 3);
            thA[kk].w[v] = *(const unsigned int*)(
                thetaT + (((size_t)b * N_ + nbase + ln) * D_ + d));
        }
    }

    v8f acc[8];
    #pragma unroll
    for (int i = 0; i < 8; ++i) acc[i] = vzero8();
    float mrow[8], lrow[8];
    #pragma unroll
    for (int i = 0; i < 8; ++i) { mrow[i] = -1e30f; lrow[i] = 0.f; }

    const int NT = N_ / 32;
    stage(0, 0);                                   // prologue

    for (int mb = 0; mb < NT; ++mb) {
        const int cur = mb & 1;
        if (mb + 1 < NT) {
            stage(1 - cur, (mb + 1) * 32);         // fill next buffer
            WAIT_ASYNCCNT(4);                      // current batch complete
        } else {
            WAIT_ASYNCCNT(0);
        }
        __syncthreads();                           // tiles visible to all waves

        // ---- S = theta * phi^T : two 16x16 tiles
        v8f S0 = vzero8(), S1 = vzero8();
        #pragma unroll
        for (int kk = 0; kk < 4; ++kk) {
            BFrag p0, p1;
            #pragma unroll
            for (int v = 0; v < 8; ++v) {
                const int d = kk * 32 + (hi ? 8 : 0) + ((v >= 4) ? 16 : 0) + 2 * (v & 3);
                p0.w[v] = *(const unsigned int*)&PhiS[cur][ln][d];
                p1.w[v] = *(const unsigned int*)&PhiS[cur][ln + 16][d];
            }
            S0 = wmma_bf16(thA[kk], p0, S0);
            S1 = wmma_bf16(thA[kk], p1, S1);
        }

        // ---- online softmax update (row reductions across 16-lane halves)
        float corr[8];
        #pragma unroll
        for (int v = 0; v < 8; ++v) {
            float mx = fmaxf(S0[v], S1[v]);
            mx = fmaxf(mx, __shfl_xor(mx, 1));
            mx = fmaxf(mx, __shfl_xor(mx, 2));
            mx = fmaxf(mx, __shfl_xor(mx, 4));
            mx = fmaxf(mx, __shfl_xor(mx, 8));
            const float mn = fmaxf(mrow[v], mx);
            corr[v] = __expf(mrow[v] - mn);
            mrow[v] = mn;
            const float p0 = __expf(S0[v] - mn);
            const float p1 = __expf(S1[v] - mn);
            float rs = p0 + p1;
            rs += __shfl_xor(rs, 1);
            rs += __shfl_xor(rs, 2);
            rs += __shfl_xor(rs, 4);
            rs += __shfl_xor(rs, 8);
            lrow[v] = lrow[v] * corr[v] + rs;
            Plds[wv][v + hi * 8][ln]      = f2bf(p0);
            Plds[wv][v + hi * 8][ln + 16] = f2bf(p1);
        }

        #pragma unroll
        for (int dt = 0; dt < 8; ++dt)
            #pragma unroll
            for (int v = 0; v < 8; ++v)
                acc[dt][v] *= corr[v];

        __builtin_amdgcn_wave_barrier();           // DS stores before DS loads

        // reload P as a 16x32 A-fragment
        BFrag pA;
        #pragma unroll
        for (int v = 0; v < 8; ++v) {
            const int k = (hi ? 8 : 0) + ((v >= 4) ? 16 : 0) + 2 * (v & 3);
            pA.w[v] = *(const unsigned int*)&Plds[wv][ln][k];
        }

        // ---- acc += P * G  (8 d-tiles of 16)
        #pragma unroll
        for (int dt = 0; dt < 8; ++dt) {
            BFrag gB;
            #pragma unroll
            for (int v = 0; v < 8; ++v) {
                const int dcol = dt * 16 + ln;
                gB.w[v] = *(const unsigned int*)&GS[cur][dcol][(hi ? 8 : 0) + ((v >= 4) ? 16 : 0) + 2 * (v & 3)];
            }
            acc[dt] = wmma_bf16(pA, gB, acc[dt]);
        }

        __syncthreads();                           // done reading buf `cur`
    }

    // ---- normalize and store y [B][N][D] bf16
    #pragma unroll
    for (int dt = 0; dt < 8; ++dt) {
        #pragma unroll
        for (int v = 0; v < 8; ++v) {
            const int nr = nbase + v + hi * 8;
            const int d  = dt * 16 + ln;
            ybf[((size_t)b * N_ + nr) * D_ + d] = f2bf(acc[dt][v] / lrow[v]);
        }
    }
}

// -----------------------------------------------------------------------------
// Kernel 3: w_y[c,n] = sum_d Ww[c,d] * y[b,n,d] + bw[c]   (M=256, K=128, N=4096)
// -----------------------------------------------------------------------------
__global__ __launch_bounds__(256)
void wgemm_kernel(const unsigned short* __restrict__ ybf,  // [B][N][D]
                  const unsigned short* __restrict__ Wwbf, // [C][D] bf16
                  const float* __restrict__ bw,
                  float* __restrict__ wy)                   // [B][C][N]
{
    const int lane  = threadIdx.x & 31;
    const int wv    = threadIdx.x >> 5;
    const int hi    = lane >> 4;
    const int ln    = lane & 15;
    const int b     = blockIdx.y;
    const int nbase = blockIdx.x * 128 + wv * 16;
    const int n     = nbase + ln;

    v8f acc[16];
    #pragma unroll
    for (int i = 0; i < 16; ++i) acc[i] = vzero8();

    for (int kk = 0; kk < 4; ++kk) {                 // K = 128, step 32
        BFrag yB;
        #pragma unroll
        for (int v = 0; v < 8; ++v) {
            const int d = kk * 32 + (hi ? 8 : 0) + ((v >= 4) ? 16 : 0) + 2 * (v & 3);
            yB.w[v] = *(const unsigned int*)(ybf + (((size_t)b * N_ + n) * D_ + d));
        }
        #pragma unroll
        for (int ct = 0; ct < 16; ++ct) {            // 16 row tiles -> c = 0..255
            BFrag wa;
            const int crow = ct * 16 + ln;
            #pragma unroll
            for (int v = 0; v < 8; ++v) {
                const int kd = kk * 32 + (hi ? 8 : 0) + ((v >= 4) ? 16 : 0) + 2 * (v & 3);
                wa.w[v] = *(const unsigned int*)(Wwbf + (size_t)crow * D_ + kd);
            }
            acc[ct] = wmma_bf16(wa, yB, acc[ct]);
        }
    }

    #pragma unroll
    for (int ct = 0; ct < 16; ++ct) {
        #pragma unroll
        for (int v = 0; v < 8; ++v) {
            const int c = ct * 16 + v + (hi ? 8 : 0);
            wy[((size_t)b * C_ + c) * N_ + n] = acc[ct][v] + bw[c];
        }
    }
}

// -----------------------------------------------------------------------------
// Kernel 4: per-channel batch statistics (deterministic tree reduction).
// -----------------------------------------------------------------------------
__global__ __launch_bounds__(256)
void stats_kernel(const float* __restrict__ wy, float* __restrict__ stats)
{
    __shared__ float ssum[256];
    __shared__ float ssq[256];
    const int c   = blockIdx.x;
    const int tid = threadIdx.x;
    float s = 0.f, q = 0.f;
    for (int i = tid; i < B_ * N_; i += 256) {
        const int bb = i / N_, nn = i % N_;
        const float v = wy[((size_t)bb * C_ + c) * N_ + nn];
        s += v;
        q += v * v;
    }
    ssum[tid] = s;
    ssq[tid]  = q;
    __syncthreads();
    for (int off = 128; off > 0; off >>= 1) {
        if (tid < off) {
            ssum[tid] += ssum[tid + off];
            ssq[tid]  += ssq[tid + off];
        }
        __syncthreads();
    }
    if (tid == 0) {
        const float inv  = 1.0f / (float)(B_ * N_);
        const float mean = ssum[0] * inv;
        const float var  = ssq[0] * inv - mean * mean;
        stats[2 * c]     = mean;
        stats[2 * c + 1] = rsqrtf(var + EPS_);
    }
}

// -----------------------------------------------------------------------------
// Kernel 5: BatchNorm affine + residual.
// -----------------------------------------------------------------------------
__global__ __launch_bounds__(256)
void bn_kernel(const float* __restrict__ wy, const float* __restrict__ stats,
               const float* __restrict__ gamma, const float* __restrict__ beta,
               const float* __restrict__ x, float* __restrict__ out)
{
    const size_t idx = (size_t)blockIdx.x * 256 + threadIdx.x;
    const int c = (int)((idx / N_) % C_);
    const float mean = stats[2 * c];
    const float istd = stats[2 * c + 1];
    out[idx] = (wy[idx] - mean) * istd * gamma[c] + beta[c] + x[idx];
}

// -----------------------------------------------------------------------------
extern "C" void kernel_launch(void* const* d_in, const int* in_sizes, int n_in,
                              void* d_out, int out_size, void* d_ws, size_t ws_size,
                              hipStream_t stream)
{
    const float* x     = (const float*)d_in[0];
    const float* g_w   = (const float*)d_in[1];
    const float* g_b   = (const float*)d_in[2];
    const float* th_w  = (const float*)d_in[3];
    const float* th_b  = (const float*)d_in[4];
    const float* ph_w  = (const float*)d_in[5];
    const float* ph_b  = (const float*)d_in[6];
    const float* w_w   = (const float*)d_in[7];
    const float* w_b   = (const float*)d_in[8];
    const float* gamma = (const float*)d_in[9];
    const float* beta  = (const float*)d_in[10];
    float* out = (float*)d_out;

    char* ws = (char*)d_ws;
    const size_t szbf = (size_t)B_ * N_ * D_ * sizeof(unsigned short); // 4 MB each
    unsigned short* thetaT = (unsigned short*)(ws);
    unsigned short* phiT   = (unsigned short*)(ws + szbf);
    unsigned short* gT     = (unsigned short*)(ws + 2 * szbf);
    unsigned short* ybf    = (unsigned short*)(ws + 3 * szbf);
    float*          wy     = (float*)(ws + 4 * szbf);                  // 16 MB
    char*           p      = ws + 4 * szbf + (size_t)B_ * C_ * N_ * sizeof(float);
    float*          stats  = (float*)p;                 p += 2 * C_ * sizeof(float);
    unsigned short* gWbf   = (unsigned short*)p;        p += (size_t)D_ * C_ * 2;
    unsigned short* tWbf   = (unsigned short*)p;        p += (size_t)D_ * C_ * 2;
    unsigned short* pWbf   = (unsigned short*)p;        p += (size_t)D_ * C_ * 2;
    unsigned short* wWbf   = (unsigned short*)p;        p += (size_t)C_ * D_ * 2;
    unsigned short* xT     = (unsigned short*)p;        // B*N*C bf16 = 8 MB

    const int wcount = D_ * C_;
    cvtw_kernel<<<dim3((wcount + 255) / 256), 256, 0, stream>>>(g_w, gWbf, wcount);
    cvtw_kernel<<<dim3((wcount + 255) / 256), 256, 0, stream>>>(th_w, tWbf, wcount);
    cvtw_kernel<<<dim3((wcount + 255) / 256), 256, 0, stream>>>(ph_w, pWbf, wcount);
    cvtw_kernel<<<dim3((wcount + 255) / 256), 256, 0, stream>>>(w_w, wWbf, wcount);
    cvtx_kernel<<<dim3((B_ * C_ * N_) / 256), 256, 0, stream>>>(x, xT);

    proj_kernel<<<dim3(N_ / 128, B_), 256, 0, stream>>>(xT, gWbf, g_b, gT, (unsigned short*)nullptr, 0);
    proj_kernel<<<dim3(N_ / 128, B_), 256, 0, stream>>>(xT, tWbf, th_b, (unsigned short*)nullptr, thetaT, 1);
    proj_kernel<<<dim3(N_ / 128, B_), 256, 0, stream>>>(xT, pWbf, ph_b, (unsigned short*)nullptr, phiT, 2);

    attn_kernel<<<dim3(N_ / 128, B_), 256, 0, stream>>>(thetaT, phiT, gT, ybf);
    wgemm_kernel<<<dim3(N_ / 128, B_), 256, 0, stream>>>(ybf, wWbf, w_b, wy);
    stats_kernel<<<dim3(C_), 256, 0, stream>>>(wy, stats);
    bn_kernel<<<dim3((B_ * C_ * N_) / 256), 256, 0, stream>>>(wy, stats, gamma, beta, x, out);
}
